// MultiAtomAttention_4432406250031
// MI455X (gfx1250) — matrix operand order repaired and checked by reading, not compiled, not run on hardware
//
#include <hip/hip_runtime.h>
#include <hip/hip_bf16.h>
#include <stdint.h>

#define NN 3072
#define DM 256
#define NH 8
#define HD 32

typedef __attribute__((ext_vector_type(16))) __bf16 v16bf;
typedef __attribute__((ext_vector_type(8)))  float  v8f;
typedef __attribute__((ext_vector_type(4)))  unsigned int v4u;
typedef __attribute__((ext_vector_type(8)))  int v8i_t;
typedef __attribute__((ext_vector_type(4)))  int v4i_t;

union V16BF { v16bf v; unsigned short u[16]; uint4 q[2]; };

__device__ __forceinline__ v8f wmma_bf16(const V16BF& a, const V16BF& b, v8f c) {
  // D = A(16x32 bf16) * B(32x16 bf16) + C(16x16 f32)
  return __builtin_amdgcn_wmma_f32_16x16x32_bf16(false, a.v, false, b.v,
                                                 (short)0, c, false, false);
}

// Load 16 bf16 as two 16-byte vector loads.
__device__ __forceinline__ V16BF ldpair(const __bf16* p0, const __bf16* p1) {
  V16BF r;
  r.q[0] = *reinterpret_cast<const uint4*>(p0);
  r.q[1] = *reinterpret_cast<const uint4*>(p1);
  return r;
}

// ---------------------------------------------------------------------------
// Tensor Data Mover: stage a [rows<=32] x 256 bf16 panel (tile 32x256) into LDS.
// D# layout per CDNA5 ISA ch.8; OOB rows (beyond tensor_dim1) zero-filled.
// amdgpu-toolchain (clang-23) 6-arg builtin:
//   (uint32x4 g0, int32x8 g1, int32x4 g2, int32x4 g3, int32x8 g4, i32 cpol)
// ---------------------------------------------------------------------------
__device__ __forceinline__ void tdm_load_panel(const __bf16* gsrc, unsigned int ldsOff,
                                               int rowsRemain) {
  unsigned long long ga = (unsigned long long)(uintptr_t)gsrc;
  v4u g0;
  g0[0] = 1u;                                            // count=1, user mode
  g0[1] = ldsOff;                                        // lds_addr (bytes)
  g0[2] = (unsigned int)(ga & 0xffffffffu);              // global_addr[31:0]
  g0[3] = (unsigned int)((ga >> 32) & 0x1ffffffu) | (2u << 30);  // addr[56:32] | type=2
  v8i_t g1;
  g1[0] = (int)(1u << 16);                               // data_size=1 (2 bytes)
  g1[1] = (int)(256u << 16);                             // tensor_dim0=256 (lo16 @ [63:48])
  g1[2] = (int)((unsigned)rowsRemain << 16);             // dim0 hi=0 | tensor_dim1 lo16
  g1[3] = (int)(((unsigned)rowsRemain >> 16) & 0xffffu)  // tensor_dim1 hi16
        | (int)(256u << 16);                             // tile_dim0=256
  g1[4] = 32;                                            // tile_dim1=32, tile_dim2=0
  g1[5] = 256;                                           // tensor_dim0_stride lo32
  g1[6] = (int)(256u << 16);                             // stride0 hi=0 | stride1 lo16
  g1[7] = 0;
  v4i_t gz4 = {0, 0, 0, 0};
  v8i_t gz8 = {0, 0, 0, 0, 0, 0, 0, 0};
  __builtin_amdgcn_tensor_load_to_lds(g0, g1, gz4, gz4, gz8, 0);
}

// ---------------------------------------------------------------------------
// Kernel 0: graph block ranges from sorted batch_index
// ---------------------------------------------------------------------------
__global__ __launch_bounds__(256) void ranges_kernel(const int* __restrict__ bidx,
                                                     int* __restrict__ gstart,
                                                     int* __restrict__ gend) {
  int i = blockIdx.x * blockDim.x + threadIdx.x;
  if (i >= NN) return;
  int b = bidx[i];
  if (i == 0 || bidx[i - 1] != b) gstart[b] = i;
  if (i == NN - 1 || bidx[i + 1] != b) gend[b] = i + 1;
}

// ---------------------------------------------------------------------------
// Kernel 0b: one-time precision prep: x -> bf16, and W matrices -> bf16
// stored TRANSPOSED so WMMA B operands are contiguous 8-bf16 runs.
// ---------------------------------------------------------------------------
__global__ __launch_bounds__(256) void prep_kernel(
    const float* __restrict__ x,
    const float* __restrict__ Wq, const float* __restrict__ Wk,
    const float* __restrict__ Wv, const float* __restrict__ Wo,
    __bf16* __restrict__ xb, __bf16* __restrict__ WqT, __bf16* __restrict__ WkT,
    __bf16* __restrict__ WvT, __bf16* __restrict__ WoT) {
  int i = blockIdx.x * 256 + threadIdx.x;
  const int NX = NN * DM;
  if (i < NX) {
    xb[i] = (__bf16)x[i];
  } else {
    int j = i - NX;                      // j < 4*DM*DM
    int m = j >> 16;                     // DM*DM == 65536
    int r = (j & 65535) >> 8;
    int c = j & 255;
    const float* W = (m == 0) ? Wq : (m == 1) ? Wk : (m == 2) ? Wv : Wo;
    __bf16* T = (m == 0) ? WqT : (m == 1) ? WkT : (m == 2) ? WvT : WoT;
    T[(size_t)c * DM + r] = (__bf16)W[(size_t)r * DM + c];
  }
}

// ---------------------------------------------------------------------------
// Kernel 1: Q/K/V projections (bf16 WMMA, f32 accum).
// One block = 16 rows; 8 waves share 3*16 (mat, col-tile) jobs.
// ---------------------------------------------------------------------------
__global__ __launch_bounds__(256) void proj_kernel(
    const __bf16* __restrict__ xb,
    const __bf16* __restrict__ WqT, const __bf16* __restrict__ WkT,
    const __bf16* __restrict__ WvT,
    __bf16* __restrict__ Qb, __bf16* __restrict__ Kb, __bf16* __restrict__ Vb) {
  const int tid = threadIdx.x;
  const int row0 = blockIdx.x * 16;
  const int wave = tid >> 5, lane = tid & 31;
  const int n15 = lane & 15;
  const int halfOff = (lane >> 4) << 3;
  const int kbA = (lane < 16) ? 0 : 8;
  const int kbB = (lane >= 16) ? 8 : 0;

  for (int job = wave; job < 48; job += 8) {
    int mat = job >> 4, ct = job & 15;
    const __bf16* WT = (mat == 0) ? WqT : ((mat == 1) ? WkT : WvT);
    __bf16* O = (mat == 0) ? Qb : ((mat == 1) ? Kb : Vb);
    v8f acc = {};
    for (int kc = 0; kc < 8; ++kc) {
      const __bf16* pa = xb + (size_t)(row0 + n15) * DM + kc * 32 + kbA;
      V16BF a = ldpair(pa, pa + 16);
      const __bf16* pb = WT + (size_t)(ct * 16 + n15) * DM + kc * 32 + kbB;
      V16BF b = ldpair(pb, pb + 16);
      acc = wmma_bf16(a, b, acc);
    }
#pragma unroll
    for (int r = 0; r < 8; ++r)
      O[(size_t)(row0 + r + halfOff) * DM + ct * 16 + n15] = (__bf16)acc[r];
  }
}

// ---------------------------------------------------------------------------
// Kernel 2: fused attention. One block = 16 rows; wave h = head h.
// Pass 1: online softmax stats (WMMA scores + shuffle reduce).
// Pass 2: weights -> LDS -> mean-over-heads output; V staged via TDM; PV WMMA.
// ---------------------------------------------------------------------------
__global__ __launch_bounds__(256) void attn_kernel(
    const __bf16* __restrict__ Qb, const __bf16* __restrict__ Kb,
    const __bf16* __restrict__ Vb,
    const float* __restrict__ rwpe, const float* __restrict__ cent,
    const float* __restrict__ pearl, const int* __restrict__ bidx,
    const int* __restrict__ gstart, const int* __restrict__ gend,
    float* __restrict__ attnMean, __bf16* __restrict__ attnO) {
  __shared__ float pHead[NH][16][32];
  __shared__ __bf16 vstage[32][DM];     // TDM-staged V panel (all heads)
  __shared__ int s_rs[16], s_re[16];

  const int tid = threadIdx.x;
  const int row0 = blockIdx.x * 16;
  const int h = tid >> 5;
  const int lane = tid & 31;
  const int n15 = lane & 15;
  const int halfOff = (lane >> 4) << 3;
  const int kbA = (lane < 16) ? 0 : 8;
  const int kbB = (lane >= 16) ? 8 : 0;
  const float SCALE = 0.17677669529663687f;   // 1/sqrt(32)
  const unsigned int stageOff = (unsigned int)(uintptr_t)(&vstage[0][0]);

  if (tid < 16) {
    int b = bidx[row0 + tid];
    s_rs[tid] = gstart[b];
    s_re[tid] = gend[b];
  }
  __syncthreads();

  int cs = s_rs[0], ce = s_re[0];
#pragma unroll
  for (int i = 1; i < 16; ++i) { cs = min(cs, s_rs[i]); ce = max(ce, s_re[i]); }
  const int cs0 = cs & ~31;
  const int cePad = (ce + 31) & ~31;
  const int ceCl = min(cePad, NN);

  // zero mean-output columns outside the active band (harness poisons d_out)
  float* meanRow = attnMean + (size_t)row0 * NN;
  for (int idx = tid; idx < 16 * NN; idx += 256) {
    int r = idx / NN, c = idx - r * NN;
    if (c < cs0 || c >= ceCl) meanRow[(size_t)r * NN + c] = 0.0f;
  }

  int rowS[8], rsS[8], reS[8];
  float rwRow[8], ctRow[8];
#pragma unroll
  for (int r = 0; r < 8; ++r) {
    int rr = r + halfOff;
    rowS[r] = row0 + rr;
    rsS[r] = s_rs[rr];
    reS[r] = s_re[rr];
    rwRow[r] = rwpe[rowS[r]];
    ctRow[r] = cent[rowS[r]];
  }

  // Q tile in A layout (constant over column chunks)
  const __bf16* qp = Qb + (size_t)(row0 + n15) * DM + h * HD + kbA;
  V16BF aQ = ldpair(qp, qp + 16);

  const float* pl = (h == 4 || h == 5) ? (pearl + (size_t)(h - 4) * NN * NN) : pearl;

  // ---- Pass 1: per-row online max / denom ----
  float mRun[8], sRun[8];
#pragma unroll
  for (int r = 0; r < 8; ++r) { mRun[r] = -__builtin_inff(); sRun[r] = 0.f; }

  for (int c0 = cs0; c0 < cePad; c0 += 32) {
    if (h >= 4 && h < 6)   // keep the pearl stream hot (global_prefetch_b8)
      __builtin_prefetch(&pl[(size_t)rowS[0] * NN + min(c0 + 32, NN - 1)], 0, 0);
#pragma unroll
    for (int t = 0; t < 2; ++t) {
      int cc = c0 + 16 * t;
      int colN = min(cc + n15, NN - 1);
      const __bf16* kp = Kb + (size_t)colN * DM + h * HD + kbB;
      V16BF bK = ldpair(kp, kp + 16);
      v8f z = {};
      v8f sc = wmma_bf16(aQ, bK, z);
      int col = cc + n15;
      int colC = min(col, NN - 1);
      float cb = (h < 2) ? rwpe[colC] : ((h < 4) ? cent[colC] : 0.f);
#pragma unroll
      for (int r = 0; r < 8; ++r) {
        float bias;
        if (h < 4)      bias = ((h < 2) ? rwRow[r] : ctRow[r]) + cb;
        else if (h < 6) bias = pl[(size_t)rowS[r] * NN + colC];
        else            bias = 0.f;
        float xv = (sc[r] + bias) * SCALE;
        bool valid = (col >= rsS[r]) && (col < reS[r]);
        if (valid) {
          if (xv > mRun[r]) { sRun[r] *= __expf(mRun[r] - xv); mRun[r] = xv; }
          sRun[r] += __expf(xv - mRun[r]);
        }
      }
    }
  }
  // combine across the 16 lanes holding the same rows
  for (int mask = 8; mask >= 1; mask >>= 1) {
#pragma unroll
    for (int r = 0; r < 8; ++r) {
      float mo = __shfl_xor(mRun[r], mask, 32);
      float so = __shfl_xor(sRun[r], mask, 32);
      float mn = fmaxf(mRun[r], mo);
      float e1 = (sRun[r] > 0.f) ? __expf(mRun[r] - mn) : 0.f;
      float e2 = (so > 0.f) ? __expf(mo - mn) : 0.f;
      sRun[r] = sRun[r] * e1 + so * e2;
      mRun[r] = mn;
    }
  }
  float inv[8];
#pragma unroll
  for (int r = 0; r < 8; ++r) inv[r] = 1.0f / sRun[r];

  // ---- Pass 2: weights, mean output, PV (V panel via Tensor Data Mover) ----
  v8f accV0 = {}, accV1 = {};
  for (int c0 = cs0; c0 < cePad; c0 += 32) {
    // wave 0 kicks the DMA for this chunk's V panel; overlaps score WMMAs
    if (h == 0)
      tdm_load_panel(Vb + (size_t)c0 * DM, stageOff, NN - c0);

#pragma unroll
    for (int t = 0; t < 2; ++t) {
      int cc = c0 + 16 * t;
      int colN = min(cc + n15, NN - 1);
      const __bf16* kp = Kb + (size_t)colN * DM + h * HD + kbB;
      V16BF bK = ldpair(kp, kp + 16);
      v8f z = {};
      v8f sc = wmma_bf16(aQ, bK, z);
      int col = cc + n15;
      int colC = min(col, NN - 1);
      float cb = (h < 2) ? rwpe[colC] : ((h < 4) ? cent[colC] : 0.f);
#pragma unroll
      for (int r = 0; r < 8; ++r) {
        float bias;
        if (h < 4)      bias = ((h < 2) ? rwRow[r] : ctRow[r]) + cb;
        else if (h < 6) bias = pl[(size_t)rowS[r] * NN + colC];
        else            bias = 0.f;
        float xv = (sc[r] + bias) * SCALE;
        bool valid = (col >= rsS[r]) && (col < reS[r]);
        float pv = valid ? __expf(xv - mRun[r]) * inv[r] : 0.f;
        pHead[h][r + halfOff][16 * t + n15] = pv;
      }
    }
    if (h == 0)
      __builtin_amdgcn_s_wait_tensorcnt(0);   // V panel resident in LDS
    __syncthreads();                          // pHead + vstage visible to all

    // mean over heads -> global
    for (int idx = tid; idx < 512; idx += 256) {
      int r = idx >> 5, c = idx & 31;
      int gc = c0 + c;
      if (gc < NN) {
        float s = 0.f;
#pragma unroll
        for (int hh = 0; hh < NH; ++hh) s += pHead[hh][r][c];
        meanRow[(size_t)r * NN + gc] = s * 0.125f;
      }
    }

    // P tile (16 rows x 32 cols) as A operand
    V16BF pA;
#pragma unroll
    for (int j = 0; j < 8; ++j) {
      pA.v[j]     = (__bf16)pHead[h][n15][kbA + j];
      pA.v[8 + j] = (__bf16)pHead[h][n15][16 + kbA + j];
    }
    // V chunk from the TDM-staged panel, two 16-wide dim halves
#pragma unroll
    for (int hf = 0; hf < 2; ++hf) {
      V16BF bV;
      int dcol = h * HD + 16 * hf + n15;
#pragma unroll
      for (int tt = 0; tt < 16; ++tt) bV.v[tt] = vstage[(tt < 8 ? kbB : 8 + kbB) + tt][dcol];
      if (hf == 0) accV0 = wmma_bf16(pA, bV, accV0);
      else         accV1 = wmma_bf16(pA, bV, accV1);
    }
    __syncthreads();   // safe to overwrite pHead / vstage next chunk
  }

  // ReLU + store bf16 attention output [row][h*32 + d]
#pragma unroll
  for (int r = 0; r < 8; ++r) {
    __bf16* o = attnO + (size_t)rowS[r] * DM + h * HD;
    o[n15]      = (__bf16)fmaxf(accV0[r], 0.f);
    o[16 + n15] = (__bf16)fmaxf(accV1[r], 0.f);
  }
}

// ---------------------------------------------------------------------------
// Kernel 3: output projection (WMMA) + bias + LayerNorm
// ---------------------------------------------------------------------------
__global__ __launch_bounds__(256) void out_kernel(
    const __bf16* __restrict__ aO, const __bf16* __restrict__ WoT,
    const float* __restrict__ bo, const float* __restrict__ gamma,
    const float* __restrict__ beta, float* __restrict__ out) {
  __shared__ float rowbuf[16][DM];
  __shared__ float ps[256], ps2[256];
  __shared__ float mu[16], rstd[16];
  const int tid = threadIdx.x;
  const int row0 = blockIdx.x * 16;
  const int wave = tid >> 5, lane = tid & 31;
  const int n15 = lane & 15;
  const int halfOff = (lane >> 4) << 3;
  const int kbA = (lane < 16) ? 0 : 8;
  const int kbB = (lane >= 16) ? 8 : 0;

  for (int ct = wave; ct < 16; ct += 8) {
    v8f acc = {};
    for (int kc = 0; kc < 8; ++kc) {
      const __bf16* pa = aO + (size_t)(row0 + n15) * DM + kc * 32 + kbA;
      V16BF a = ldpair(pa, pa + 16);
      const __bf16* pb = WoT + (size_t)(ct * 16 + n15) * DM + kc * 32 + kbB;
      V16BF b = ldpair(pb, pb + 16);
      acc = wmma_bf16(a, b, acc);
    }
    int col = ct * 16 + n15;
    float bb = bo[col];
#pragma unroll
    for (int r = 0; r < 8; ++r) rowbuf[r + halfOff][col] = acc[r] + bb;
  }
  __syncthreads();

  int r = tid >> 4, part = tid & 15;
  float s = 0.f, s2 = 0.f;
#pragma unroll
  for (int i = 0; i < 16; ++i) {
    float v = rowbuf[r][part * 16 + i];
    s += v; s2 += v * v;
  }
  ps[tid] = s; ps2[tid] = s2;
  __syncthreads();
  if (part == 0) {
    float S = 0.f, S2 = 0.f;
#pragma unroll
    for (int i = 0; i < 16; ++i) { S += ps[r * 16 + i]; S2 += ps2[r * 16 + i]; }
    float m = S * (1.0f / DM);
    float var = S2 * (1.0f / DM) - m * m;
    mu[r] = m;
    rstd[r] = rsqrtf(var + 1e-5f);
  }
  __syncthreads();
#pragma unroll
  for (int i = 0; i < 16; ++i) {
    int col = part * 16 + i;
    float v = (rowbuf[r][col] - mu[r]) * rstd[r] * gamma[col] + beta[col];
    out[(size_t)(row0 + r) * DM + col] = v;
  }
}

// ---------------------------------------------------------------------------
extern "C" void kernel_launch(void* const* d_in, const int* in_sizes, int n_in,
                              void* d_out, int out_size, void* d_ws, size_t ws_size,
                              hipStream_t stream) {
  (void)in_sizes; (void)n_in; (void)out_size; (void)ws_size;
  const float* x     = (const float*)d_in[0];
  const float* rwpe  = (const float*)d_in[1];
  const float* cent  = (const float*)d_in[2];
  const float* pearl = (const float*)d_in[3];
  const int*   bidx  = (const int*)d_in[4];
  const float* Wq    = (const float*)d_in[5];
  const float* Wk    = (const float*)d_in[6];
  const float* Wv    = (const float*)d_in[7];
  const float* Wo    = (const float*)d_in[8];
  const float* bo    = (const float*)d_in[9];
  const float* gamma = (const float*)d_in[10];
  const float* beta  = (const float*)d_in[11];

  const size_t NDM = (size_t)NN * DM;
  const size_t DD  = (size_t)DM * DM;
  __bf16* xb  = (__bf16*)d_ws;
  __bf16* Qb  = xb + NDM;
  __bf16* Kb  = Qb + NDM;
  __bf16* Vb  = Kb + NDM;
  __bf16* aO  = Vb + NDM;
  __bf16* WqT = aO + NDM;
  __bf16* WkT = WqT + DD;
  __bf16* WvT = WkT + DD;
  __bf16* WoT = WvT + DD;
  int* gstart = (int*)(WoT + DD);
  int* gend   = gstart + 64;

  float* outMain  = (float*)d_out;
  float* attnMean = outMain + NDM;

  hipLaunchKernelGGL(ranges_kernel, dim3(NN / 256), dim3(256), 0, stream,
                     bidx, gstart, gend);
  hipLaunchKernelGGL(prep_kernel, dim3((NN * DM + 4 * DM * DM) / 256), dim3(256), 0,
                     stream, x, Wq, Wk, Wv, Wo, xb, WqT, WkT, WvT, WoT);
  hipLaunchKernelGGL(proj_kernel, dim3(NN / 16), dim3(256), 0, stream,
                     xb, WqT, WkT, WvT, Qb, Kb, Vb);
  hipLaunchKernelGGL(attn_kernel, dim3(NN / 16), dim3(256), 0, stream,
                     Qb, Kb, Vb, rwpe, cent, pearl, bidx, gstart, gend,
                     attnMean, aO);
  hipLaunchKernelGGL(out_kernel, dim3(NN / 16), dim3(256), 0, stream,
                     aO, WoT, bo, gamma, beta, outMain);
}
